// RicciFlowGNN_282
// MI455X (gfx1250) — matrix operand order repaired and checked by reading, not compiled, not run on hardware
//
#include <hip/hip_runtime.h>
#include <math.h>

#define N_NODES 100000
#define N_EDGES 1600000

typedef __bf16 bf16_t;
typedef bf16_t v16bf __attribute__((ext_vector_type(16)));
typedef float  v8f   __attribute__((ext_vector_type(8)));

struct alignas(16) B8 { bf16_t v[8]; };

__device__ __forceinline__ unsigned short f2bfu(float f) {
  unsigned u = __float_as_uint(f);
  u += 0x7FFFu + ((u >> 16) & 1u);
  return (unsigned short)(u >> 16);
}

// Branchless fast activations: v_exp_f32 + v_rcp_f32, correct saturation.
__device__ __forceinline__ float fast_tanh(float x) {
  float t = __expf(2.f * x);                    // inf for large x, 0 for very neg
  return 1.f - 2.f * __builtin_amdgcn_rcpf(t + 1.f);
}
__device__ __forceinline__ float fast_sigmoid(float x) {
  return __builtin_amdgcn_rcpf(1.f + __expf(-x));
}

__device__ __forceinline__ v16bf mkfrag(B8 c0, B8 c1) {
  v16bf a;
#pragma unroll
  for (int i = 0; i < 8; ++i) { a[i] = c0.v[i]; a[i + 8] = c1.v[i]; }
  return a;
}

// A fragment (16 rows x 32 K, bf16). p = row base + k-tile*32 elements.
// Lane half picks K {0..7,16..23} vs {8..15,24..31} per ISA A layout.
__device__ __forceinline__ v16bf ldA(const unsigned short* p, int lane) {
  int off = (lane >> 4) * 8;
  B8 c0 = *(const B8*)(p + off);
  B8 c1 = *(const B8*)(p + off + 16);
  return mkfrag(c0, c1);
}

// B fragment from LDS weights pre-swizzled as [kt][nt][lane][16].
__device__ __forceinline__ v16bf ldB(const unsigned short* w, int kt, int nt, int lane) {
  const unsigned short* p = w + ((((kt * 4 + nt) * 32) + lane) << 4);
  B8 c0 = *(const B8*)(p);
  B8 c1 = *(const B8*)(p + 8);
  return mkfrag(c0, c1);
}

__device__ __forceinline__ v8f wmma_bf16(v16bf a, v16bf b, v8f c) {
  return __builtin_amdgcn_wmma_f32_16x16x32_bf16(false, a, false, b, (short)0, c, false, false);
}

__device__ __forceinline__ float hredux16(float v) {
#pragma unroll
  for (int m = 1; m < 16; m <<= 1) v += __shfl_xor(v, m, 32);
  return v;
}
__device__ __forceinline__ float redux32(float v) {
#pragma unroll
  for (int m = 1; m < 32; m <<= 1) v += __shfl_xor(v, m, 32);
  return v;
}

// Swizzle f32 weight matrix W[K][64] into bf16 B-fragment layout in LDS.
__device__ __forceinline__ void fill_swz(unsigned short* dst, const float* W, int K,
                                         int tid, int nthr) {
  int total = K * 64;
  for (int idx = tid; idx < total; idx += nthr) {
    int j = idx & 15;
    int l = (idx >> 4) & 31;
    int nt = (idx >> 9) & 3;
    int kt = idx >> 11;
    int k = kt * 32 + (l >> 4) * 8 + (j & 7) + (j >> 3) * 16;
    int n = nt * 16 + (l & 15);
    dst[idx] = f2bfu(W[k * 64 + n]);
  }
}

// ---------------- h = x @ in_w + in_b -------------------------------------
__global__ __launch_bounds__(256) void k_input(const float* __restrict__ x,
                                               const float* __restrict__ in_w,
                                               const float* __restrict__ in_b,
                                               float* __restrict__ h,
                                               unsigned short* __restrict__ hbf) {
  __shared__ float sW[8 * 64];
  __shared__ float sB[64];
  int tid = threadIdx.x;
  for (int i = tid; i < 8 * 64; i += 256) sW[i] = in_w[i];
  if (tid < 64) sB[tid] = in_b[tid];
  __syncthreads();
  int n = blockIdx.x * 256 + tid;
  if (n >= N_NODES) return;
  float xv[8];
#pragma unroll
  for (int f = 0; f < 8; ++f) xv[f] = x[(size_t)n * 8 + f];
  for (int j = 0; j < 64; ++j) {
    float s = sB[j];
#pragma unroll
    for (int f = 0; f < 8; ++f) s += xv[f] * sW[f * 64 + j];
    h[(size_t)n * 64 + j] = s;
    hbf[(size_t)n * 64 + j] = f2bfu(s);
  }
}

// ---------------- gate + msg + scatter (WMMA) ------------------------------
__global__ __launch_bounds__(256) void k_edge_msg(
    const long long* __restrict__ erow, const long long* __restrict__ ecol,
    const float* __restrict__ ricci, const unsigned short* __restrict__ hbf,
    float* __restrict__ aggr,
    const float* gw1, const float* gb1, const float* gw2, const float* gb2,
    const float* mw1, const float* mb1, const float* mw2, const float* mb2) {
  __shared__ alignas(16) unsigned short sGW1[128 * 64];
  __shared__ alignas(16) unsigned short sMW1[64 * 64];
  __shared__ alignas(16) unsigned short sMW2[64 * 64];
  __shared__ alignas(16) unsigned short sHid[8][16 * 64];
  __shared__ float sGW1r[64], sMW1r[64], sGB1[64], sMB1[64], sMB2[64], sGW2[64];
  __shared__ unsigned sColIdx[8][16];
  __shared__ float sRf[8][16], sGate[8][16];
  __shared__ float sGB2s;

  int tid = threadIdx.x;
  fill_swz(sGW1, gw1, 128, tid, 256);
  fill_swz(sMW1, mw1, 64, tid, 256);
  fill_swz(sMW2, mw2, 64, tid, 256);
  if (tid < 64) {
    sGW1r[tid] = gw1[128 * 64 + tid];   // ricci row of gate_w1
    sMW1r[tid] = mw1[64 * 64 + tid];    // ricci row of msg_w1
    sGB1[tid] = gb1[tid];
    sMB1[tid] = mb1[tid];
    sMB2[tid] = mb2[tid];
    sGW2[tid] = gw2[tid];
  }
  if (tid == 0) sGB2s = gb2[0];

  int wave = tid >> 5, lane = tid & 31;
  long long tileBase = ((long long)blockIdx.x * 8 + wave) * 16;
  int m = lane & 15;
  unsigned myR = (unsigned)erow[tileBase + m];   // x_j = h[row]
  unsigned myC = (unsigned)ecol[tileBase + m];   // x_i = h[col]
  if (lane < 16) {
    sColIdx[wave][lane] = myC;
    sRf[wave][lane] = ricci[tileBase + lane];
  }
  __syncthreads();

  const unsigned short* pI = hbf + (size_t)myC * 64;
  const unsigned short* pJ = hbf + (size_t)myR * 64;
  v16bf aI0 = ldA(pI, lane), aI1 = ldA(pI + 32, lane);
  v16bf aJ0 = ldA(pJ, lane), aJ1 = ldA(pJ + 32, lane);

  // ---- gate: sigmoid(tanh([x_i|x_j|rf] @ W1 + b1) @ W2 + b2) ----
  float gs[8];
#pragma unroll
  for (int r = 0; r < 8; ++r) gs[r] = 0.f;
#pragma unroll
  for (int nt = 0; nt < 4; ++nt) {
    v8f acc = {};
    acc = wmma_bf16(aI0, ldB(sGW1, 0, nt, lane), acc);
    acc = wmma_bf16(aI1, ldB(sGW1, 1, nt, lane), acc);
    acc = wmma_bf16(aJ0, ldB(sGW1, 2, nt, lane), acc);
    acc = wmma_bf16(aJ1, ldB(sGW1, 3, nt, lane), acc);
    int nc = nt * 16 + (lane & 15);
    float wr = sGW1r[nc], bb = sGB1[nc], w2 = sGW2[nc];
#pragma unroll
    for (int r = 0; r < 8; ++r) {
      int mr = (lane >> 4) * 8 + r;
      float v = acc[r] + sRf[wave][mr] * wr + bb;
      gs[r] += fast_tanh(v) * w2;
    }
  }
#pragma unroll
  for (int r = 0; r < 8; ++r) gs[r] = hredux16(gs[r]);
  if ((lane & 15) == 0) {
#pragma unroll
    for (int r = 0; r < 8; ++r) {
      int mr = (lane >> 4) * 8 + r;
      sGate[wave][mr] = fast_sigmoid(gs[r] + sGB2s);
    }
  }

  // ---- msg layer 1: relu([x_j|rf] @ W1 + b1) -> LDS (bf16) ----
#pragma unroll
  for (int nt = 0; nt < 4; ++nt) {
    v8f acc = {};
    acc = wmma_bf16(aJ0, ldB(sMW1, 0, nt, lane), acc);
    acc = wmma_bf16(aJ1, ldB(sMW1, 1, nt, lane), acc);
    int nc = nt * 16 + (lane & 15);
    float wr = sMW1r[nc], bb = sMB1[nc];
#pragma unroll
    for (int r = 0; r < 8; ++r) {
      int mr = (lane >> 4) * 8 + r;
      float v = acc[r] + sRf[wave][mr] * wr + bb;
      sHid[wave][mr * 64 + nc] = f2bfu(fmaxf(v, 0.f));
    }
  }
  __syncthreads();

  // ---- msg layer 2 + gate scale + scatter-add to aggr[col] ----
  const unsigned short* ph = &sHid[wave][(lane & 15) * 64];
  v16bf h0 = ldA(ph, lane), h1 = ldA(ph + 32, lane);
#pragma unroll
  for (int nt = 0; nt < 4; ++nt) {
    v8f acc = {};
    acc = wmma_bf16(h0, ldB(sMW2, 0, nt, lane), acc);
    acc = wmma_bf16(h1, ldB(sMW2, 1, nt, lane), acc);
    int nc = nt * 16 + (lane & 15);
    float bb = sMB2[nc];
#pragma unroll
    for (int r = 0; r < 8; ++r) {
      int mr = (lane >> 4) * 8 + r;
      float val = (acc[r] + bb) * sGate[wave][mr];
      atomicAdd(&aggr[(size_t)sColIdx[wave][mr] * 64 + nc], val);
    }
  }
}

// ---------------- node update: h = LN(h + relu([h|aggr]@W + b)) ------------
__global__ __launch_bounds__(256) void k_node_update(
    float* __restrict__ h, unsigned short* __restrict__ hbf,
    const float* __restrict__ aggr,
    const float* uw, const float* ub, const float* lng, const float* lnb) {
  __shared__ float sW[128 * 64];
  __shared__ float sB[64], sG[64], sLB[64];
  __shared__ float sComb[8][128];
  int tid = threadIdx.x;
  for (int i = tid; i < 128 * 64; i += 256) sW[i] = uw[i];
  if (tid < 64) { sB[tid] = ub[tid]; sG[tid] = lng[tid]; sLB[tid] = lnb[tid]; }
  __syncthreads();
  int wave = tid >> 5, lane = tid & 31;
  int node = blockIdx.x * 8 + wave;
  size_t base = (size_t)node * 64;
  sComb[wave][lane]      = h[base + lane];
  sComb[wave][lane + 32] = h[base + lane + 32];
  sComb[wave][64 + lane]      = aggr[base + lane];
  sComb[wave][96 + lane]      = aggr[base + lane + 32];
  __syncthreads();
  float a0 = sB[lane], a1 = sB[lane + 32];
  for (int k = 0; k < 128; ++k) {
    float c = sComb[wave][k];
    a0 += c * sW[k * 64 + lane];
    a1 += c * sW[k * 64 + lane + 32];
  }
  float u0 = sComb[wave][lane] + fmaxf(a0, 0.f);
  float u1 = sComb[wave][lane + 32] + fmaxf(a1, 0.f);
  float s  = redux32(u0 + u1);
  float s2 = redux32(u0 * u0 + u1 * u1);
  float mu = s * (1.f / 64.f);
  float var = s2 * (1.f / 64.f) - mu * mu;
  float inv = rsqrtf(var + 1e-5f);
  float y0 = (u0 - mu) * inv * sG[lane] + sLB[lane];
  float y1 = (u1 - mu) * inv * sG[lane + 32] + sLB[lane + 32];
  h[base + lane] = y0;  h[base + lane + 32] = y1;
  hbf[base + lane] = f2bfu(y0);  hbf[base + lane + 32] = f2bfu(y1);
}

// ---------------- curvature update (WMMA) ----------------------------------
__global__ __launch_bounds__(256) void k_curv(
    const long long* __restrict__ erow, const long long* __restrict__ ecol,
    const unsigned short* __restrict__ hbf, float* __restrict__ ricci,
    const float* cw1, const float* cb1, const float* cw2, const float* cb2) {
  __shared__ alignas(16) unsigned short sW1[128 * 64];
  __shared__ float sB1[64], sW2[64];
  __shared__ float sCB2;
  int tid = threadIdx.x;
  fill_swz(sW1, cw1, 128, tid, 256);
  if (tid < 64) { sB1[tid] = cb1[tid]; sW2[tid] = cw2[tid]; }
  if (tid == 0) sCB2 = cb2[0];
  __syncthreads();
  int wave = tid >> 5, lane = tid & 31;
  long long tileBase = ((long long)blockIdx.x * 8 + wave) * 16;
  int m = lane & 15;
  unsigned myR = (unsigned)erow[tileBase + m];
  unsigned myC = (unsigned)ecol[tileBase + m];
  const unsigned short* pR = hbf + (size_t)myR * 64;
  const unsigned short* pC = hbf + (size_t)myC * 64;
  v16bf a0 = ldA(pR, lane), a1 = ldA(pR + 32, lane);
  v16bf a2 = ldA(pC, lane), a3 = ldA(pC + 32, lane);
  float gs[8];
#pragma unroll
  for (int r = 0; r < 8; ++r) gs[r] = 0.f;
#pragma unroll
  for (int nt = 0; nt < 4; ++nt) {
    v8f acc = {};
    acc = wmma_bf16(a0, ldB(sW1, 0, nt, lane), acc);
    acc = wmma_bf16(a1, ldB(sW1, 1, nt, lane), acc);
    acc = wmma_bf16(a2, ldB(sW1, 2, nt, lane), acc);
    acc = wmma_bf16(a3, ldB(sW1, 3, nt, lane), acc);
    int nc = nt * 16 + (lane & 15);
    float bb = sB1[nc], w2 = sW2[nc];
#pragma unroll
    for (int r = 0; r < 8; ++r) gs[r] += fast_tanh(acc[r] + bb) * w2;
  }
#pragma unroll
  for (int r = 0; r < 8; ++r) gs[r] = hredux16(gs[r]);
  if ((lane & 15) == 0) {
#pragma unroll
    for (int r = 0; r < 8; ++r) {
      long long e = tileBase + (lane >> 4) * 8 + r;
      float upd = fast_tanh(gs[r] + sCB2);
      ricci[e] = 0.7f * ricci[e] + 0.3f * upd;
    }
  }
}

// ---------------- edge weight head (WMMA) ----------------------------------
__global__ __launch_bounds__(256) void k_ew(
    const long long* __restrict__ erow, const long long* __restrict__ ecol,
    const unsigned short* __restrict__ hbf, const float* __restrict__ ricci,
    float* __restrict__ ew,
    const float* w1, const float* b1, const float* w2, const float* b2) {
  __shared__ alignas(16) unsigned short sW1[128 * 64];
  __shared__ float sW1r[64], sB1[64], sW2[64];
  __shared__ float sRf[8][16];
  __shared__ float sEB2;
  int tid = threadIdx.x;
  fill_swz(sW1, w1, 128, tid, 256);
  if (tid < 64) { sW1r[tid] = w1[128 * 64 + tid]; sB1[tid] = b1[tid]; sW2[tid] = w2[tid]; }
  if (tid == 0) sEB2 = b2[0];
  int wave = tid >> 5, lane = tid & 31;
  long long tileBase = ((long long)blockIdx.x * 8 + wave) * 16;
  int m = lane & 15;
  unsigned myR = (unsigned)erow[tileBase + m];
  unsigned myC = (unsigned)ecol[tileBase + m];
  if (lane < 16) sRf[wave][lane] = ricci[tileBase + lane];
  __syncthreads();
  const unsigned short* pR = hbf + (size_t)myR * 64;
  const unsigned short* pC = hbf + (size_t)myC * 64;
  v16bf a0 = ldA(pR, lane), a1 = ldA(pR + 32, lane);
  v16bf a2 = ldA(pC, lane), a3 = ldA(pC + 32, lane);
  float gs[8];
#pragma unroll
  for (int r = 0; r < 8; ++r) gs[r] = 0.f;
#pragma unroll
  for (int nt = 0; nt < 4; ++nt) {
    v8f acc = {};
    acc = wmma_bf16(a0, ldB(sW1, 0, nt, lane), acc);
    acc = wmma_bf16(a1, ldB(sW1, 1, nt, lane), acc);
    acc = wmma_bf16(a2, ldB(sW1, 2, nt, lane), acc);
    acc = wmma_bf16(a3, ldB(sW1, 3, nt, lane), acc);
    int nc = nt * 16 + (lane & 15);
    float wr = sW1r[nc], bb = sB1[nc], w2c = sW2[nc];
#pragma unroll
    for (int r = 0; r < 8; ++r) {
      int mr = (lane >> 4) * 8 + r;
      float v = acc[r] + sRf[wave][mr] * wr + bb;
      gs[r] += fmaxf(v, 0.f) * w2c;
    }
  }
#pragma unroll
  for (int r = 0; r < 8; ++r) gs[r] = hredux16(gs[r]);
  if ((lane & 15) == 0) {
#pragma unroll
    for (int r = 0; r < 8; ++r) {
      long long e = tileBase + (lane >> 4) * 8 + r;
      ew[e] = fast_sigmoid(gs[r] + sEB2);
    }
  }
}

// ---------------- node risk + graph-sum accumulation -----------------------
__global__ __launch_bounds__(256) void k_node_final(
    const float* __restrict__ h, float* __restrict__ risk, float* __restrict__ gsum,
    const float* w1, const float* b1, const float* w2, const float* b2) {
  __shared__ float sW1[64 * 32];
  __shared__ float sB1[32], sW2[32];
  __shared__ float sH[8][64];
  __shared__ float sB2;
  int tid = threadIdx.x;
  for (int i = tid; i < 64 * 32; i += 256) sW1[i] = w1[i];
  if (tid < 32) { sB1[tid] = b1[tid]; sW2[tid] = w2[tid]; }
  if (tid == 0) sB2 = b2[0];
  __syncthreads();
  int wave = tid >> 5, lane = tid & 31;
  int node = blockIdx.x * 8 + wave;
  size_t base = (size_t)node * 64;
  float h0 = h[base + lane], h1 = h[base + lane + 32];
  sH[wave][lane] = h0;  sH[wave][lane + 32] = h1;
  atomicAdd(&gsum[lane], h0);
  atomicAdd(&gsum[lane + 32], h1);
  __syncthreads();
  float a = sB1[lane];
  for (int k = 0; k < 64; ++k) a += sH[wave][k] * sW1[k * 32 + lane];
  a = fmaxf(a, 0.f);
  float sacc = redux32(a * sW2[lane]);
  if (lane == 0) risk[node] = fast_sigmoid(sacc + sB2);
}

// ---------------- graph summary head ---------------------------------------
__global__ void k_gs(const float* __restrict__ gsum, float* __restrict__ out,
                     const float* w1, const float* b1, const float* w2, const float* b2) {
  __shared__ float hid[32];
  int t = threadIdx.x;
  if (t < 32) {
    float v = b1[t];
    for (int k = 0; k < 64; ++k) v += (gsum[k] * (1.f / (float)N_NODES)) * w1[k * 32 + t];
    hid[t] = fmaxf(v, 0.f);
  }
  __syncthreads();
  if (t < 3) {
    float o = b2[t];
    for (int k = 0; k < 32; ++k) o += hid[k] * w2[k * 3 + t];
    out[t] = o;
  }
}

extern "C" void kernel_launch(void* const* d_in, const int* in_sizes, int n_in,
                              void* d_out, int out_size, void* d_ws, size_t ws_size,
                              hipStream_t stream) {
  (void)in_sizes; (void)n_in; (void)out_size; (void)ws_size;
  const float*      x        = (const float*)d_in[0];
  const long long*  eidx     = (const long long*)d_in[1];
  const float*      ricci_in = (const float*)d_in[2];
  // params pytree leaves (dict keys sorted):
  const float* ew_b1 = (const float*)d_in[3];
  const float* ew_b2 = (const float*)d_in[4];
  const float* ew_w1 = (const float*)d_in[5];
  const float* ew_w2 = (const float*)d_in[6];
  const float* gs_b1 = (const float*)d_in[7];
  const float* gs_b2 = (const float*)d_in[8];
  const float* gs_w1 = (const float*)d_in[9];
  const float* gs_w2 = (const float*)d_in[10];
  const float* in_b  = (const float*)d_in[11];
  const float* in_w  = (const float*)d_in[12];
  const float* nr_b1 = (const float*)d_in[61];
  const float* nr_b2 = (const float*)d_in[62];
  const float* nr_w1 = (const float*)d_in[63];
  const float* nr_w2 = (const float*)d_in[64];

  float* out_h     = (float*)d_out;
  float* out_ew    = out_h + (size_t)N_NODES * 64;
  float* out_nr    = out_ew + N_EDGES;
  float* out_gs    = out_nr + N_NODES;
  float* out_ricci = out_gs + 3;

  float*          aggr = (float*)d_ws;
  unsigned short* hbf  = (unsigned short*)((char*)d_ws + (size_t)N_NODES * 64 * 4);
  float*          gsum = (float*)((char*)d_ws + (size_t)N_NODES * 64 * 6);

  const long long* erow = eidx;
  const long long* ecol = eidx + N_EDGES;

  hipMemcpyAsync(out_ricci, ricci_in, (size_t)N_EDGES * sizeof(float),
                 hipMemcpyDeviceToDevice, stream);
  k_input<<<(N_NODES + 255) / 256, 256, 0, stream>>>(x, in_w, in_b, out_h, hbf);

  const int egrid = N_EDGES / 128;  // 8 waves x 16 edges per block
  const int ngrid = N_NODES / 8;    // 8 waves x 1 node per block

  for (int l = 0; l < 3; ++l) {
    const int pb = 13 + 16 * l;
    const float* curv_b1 = (const float*)d_in[pb + 0];
    const float* curv_b2 = (const float*)d_in[pb + 1];
    const float* curv_w1 = (const float*)d_in[pb + 2];
    const float* curv_w2 = (const float*)d_in[pb + 3];
    const float* gate_b1 = (const float*)d_in[pb + 4];
    const float* gate_b2 = (const float*)d_in[pb + 5];
    const float* gate_w1 = (const float*)d_in[pb + 6];
    const float* gate_w2 = (const float*)d_in[pb + 7];
    const float* ln_b    = (const float*)d_in[pb + 8];
    const float* ln_g    = (const float*)d_in[pb + 9];
    const float* msg_b1  = (const float*)d_in[pb + 10];
    const float* msg_b2  = (const float*)d_in[pb + 11];
    const float* msg_w1  = (const float*)d_in[pb + 12];
    const float* msg_w2  = (const float*)d_in[pb + 13];
    const float* upd_b   = (const float*)d_in[pb + 14];
    const float* upd_w   = (const float*)d_in[pb + 15];

    hipMemsetAsync(aggr, 0, (size_t)N_NODES * 64 * sizeof(float), stream);
    k_edge_msg<<<egrid, 256, 0, stream>>>(erow, ecol, out_ricci, hbf, aggr,
        gate_w1, gate_b1, gate_w2, gate_b2, msg_w1, msg_b1, msg_w2, msg_b2);
    k_node_update<<<ngrid, 256, 0, stream>>>(out_h, hbf, aggr, upd_w, upd_b, ln_g, ln_b);
    k_curv<<<egrid, 256, 0, stream>>>(erow, ecol, hbf, out_ricci,
        curv_w1, curv_b1, curv_w2, curv_b2);
  }

  k_ew<<<egrid, 256, 0, stream>>>(erow, ecol, hbf, out_ricci, out_ew,
      ew_w1, ew_b1, ew_w2, ew_b2);
  hipMemsetAsync(gsum, 0, 64 * sizeof(float), stream);
  k_node_final<<<ngrid, 256, 0, stream>>>(out_h, out_nr, gsum, nr_w1, nr_b1, nr_w2, nr_b2);
  k_gs<<<1, 64, 0, stream>>>(gsum, out_gs, gs_w1, gs_b1, gs_w2, gs_b2);
}
